// SinogramToPatchesConnected_8254927143152
// MI455X (gfx1250) — matrix operand order repaired and checked by reading, not compiled, not run on hardware
//
#include <hip/hip_runtime.h>
#include <hip/hip_bf16.h>
#include <math.h>

typedef __attribute__((ext_vector_type(16))) _Float16 v16h;
typedef __attribute__((ext_vector_type(8)))  _Float16 v8h;
typedef __attribute__((ext_vector_type(8)))  float    v8f;

#define IMG      128
#define NANG     60
#define NPIX     7680          // 60*128
#define NPATCH   625
#define TILE_H   6             // conv3 output rows per tile
#define NTILES   10            // 60 / 6
#define A1_ROWS  10            // TILE_H + 4 (conv1 halo)
#define A2_ROWS  8             // TILE_H + 2 (conv2 halo)
#define A1_W     130           // 128 cols + 2 zero-pad cols
#define A2_W     130
#define A1_ROW_STRIDE (A1_W*32)        // halves per a1 row
#define A2_CH_STRIDE  (A2_ROWS*A2_W)   // halves per a2 channel plane

#define A1_HALVES (A1_ROWS*A1_ROW_STRIDE)   // 41600
#define A2_HALVES (64*A2_CH_STRIDE)         // 66560
#define W2_HALVES (9*64*32)                 // 18432
#define HALVES_TOTAL (A1_HALVES+A2_HALVES+W2_HALVES)
#define NFLOATS (288+32+64+576+200)         // w1,b1,b2,w3,red
#define SMEM_BYTES (HALVES_TOTAL*2 + NFLOATS*4)   // 257,824 B (< 320KB/WGP)

// ---------------------------------------------------------------------------
// Fused per-batch kernel: masked sinogram -> conv1(relu) -> conv2(WMMA,relu)
// -> conv3 -> linear(7680->25), all through LDS, one block per batch image.
// ---------------------------------------------------------------------------
__global__ __launch_bounds__(256) void fused_batch(
    const float* __restrict__ sino,
    const float* __restrict__ w1, const float* __restrict__ b1,
    const float* __restrict__ w2, const float* __restrict__ b2,
    const float* __restrict__ w3, const float* __restrict__ b3,
    const float* __restrict__ linw, const float* __restrict__ linb,
    const int*   __restrict__ masks,
    float* __restrict__ patches)
{
    extern __shared__ char smem[];
    _Float16* s_a1 = (_Float16*)smem;                 // [10][130][32] f16
    _Float16* s_a2 = s_a1 + A1_HALVES;                // [64][8][130]  f16
    _Float16* s_w2 = s_a2 + A2_HALVES;                // [9][64][32]   f16
    float* s_f   = (float*)(s_w2 + W2_HALVES);
    float* s_w1  = s_f;                               // 288
    float* s_b1  = s_w1 + 288;                        // 32
    float* s_b2  = s_b1 + 32;                         // 64
    float* s_w3  = s_b2 + 64;                         // 576
    float* s_red = s_w3 + 576;                        // 8*25

    const int p    = blockIdx.x;
    const int tid  = threadIdx.x;
    const int lane = tid & 31;
    const int wave = tid >> 5;
    const int nn   = lane & 15;      // M for A-side, N for B/D-side
    const bool hi  = lane >= 16;

    // ---- stage weights into LDS (B-fragment friendly layout for conv2) ----
    for (int j = tid; j < W2_HALVES; j += 256) {
        int t = j >> 11;             // tap 0..8
        int o = (j >> 5) & 63;       // out channel
        int k = j & 31;              // in channel
        s_w2[j] = (_Float16)w2[o*288 + k*9 + t];   // OIHW flat
    }
    for (int j = tid; j < 288; j += 256) s_w1[j] = w1[j];
    if (tid < 32) s_b1[tid] = b1[tid];
    if (tid < 64) s_b2[tid] = b2[tid];
    for (int j = tid; j < 576; j += 256) s_w3[j] = w3[j];
    __syncthreads();

    const int* maskp = masks + p*NPIX;
    const float cb3 = b3[0];
    float linacc[25];
    #pragma unroll
    for (int q = 0; q < 25; ++q) linacc[q] = 0.f;

    for (int tile = 0; tile < NTILES; ++tile) {
        const int r0 = tile * TILE_H;

        // ---------- conv1 (scalar): rows r0-2 .. r0+7 into s_a1 ----------
        for (int j = tid; j < A1_ROWS*32; j += 256) {   // zero pad cols 0,129
            int rr = j >> 5, ch = j & 31;
            s_a1[rr*A1_ROW_STRIDE + ch] = (_Float16)0.f;
            s_a1[rr*A1_ROW_STRIDE + 129*32 + ch] = (_Float16)0.f;
        }
        for (int i = tid; i < A1_ROWS*IMG; i += 256) {  // 1280 sites (uniform)
            int a1r = i >> 7;
            int c   = i & 127;
            int gr  = r0 - 2 + a1r;
            _Float16* dst = s_a1 + a1r*A1_ROW_STRIDE + (c+1)*32;
            if (gr < 0 || gr >= NANG) {
                #pragma unroll
                for (int ch = 0; ch < 32; ++ch) dst[ch] = (_Float16)0.f;
            } else {
                float v[9];
                #pragma unroll
                for (int dy = 0; dy < 3; ++dy)
                    #pragma unroll
                    for (int dx = 0; dx < 3; ++dx) {
                        int rr = gr + dy - 1, cc = c + dx - 1;
                        float mv = 0.f;
                        if (rr >= 0 && rr < NANG && cc >= 0 && cc < IMG) {
                            int ii = rr*IMG + cc;
                            mv = maskp[ii] ? sino[ii] : 0.f;
                        }
                        v[dy*3+dx] = mv;
                    }
                #pragma unroll
                for (int ch = 0; ch < 32; ++ch) {
                    float s = s_b1[ch];
                    #pragma unroll
                    for (int t = 0; t < 9; ++t) s = fmaf(v[t], s_w1[ch*9+t], s);
                    dst[ch] = (_Float16)fmaxf(s, 0.f);
                }
            }
        }
        for (int j = tid; j < 64*A2_ROWS; j += 256) {   // zero s_a2 pad cols
            int ch = j >> 3, rr = j & 7;
            s_a2[ch*A2_CH_STRIDE + rr*A2_W + 0]   = (_Float16)0.f;
            s_a2[ch*A2_CH_STRIDE + rr*A2_W + 129] = (_Float16)0.f;
        }
        __syncthreads();

        // ---------- conv2 via WMMA: 8 rows x 128 cols = 64 M-tiles ----------
        // Per tap: K=32 in-channels == one v_wmma_f32_16x16x32_f16 K-step.
        for (int mt = wave; mt < 64; mt += 8) {
            const int a2r     = mt >> 3;
            const int colbase = (mt & 7) << 4;
            const int gr      = r0 - 1 + a2r;
            const bool rowvalid = (gr >= 0) && (gr < NANG);
            if (rowvalid) {
                v8f acc[4];
                #pragma unroll
                for (int n = 0; n < 4; ++n) {
                    v8f z = {0.f,0.f,0.f,0.f,0.f,0.f,0.f,0.f};
                    acc[n] = z;
                }
                const int kbA = hi ? 8 : 0;    // A K-layout per ISA
                const int kbB = hi ? 8 : 0;    // B K-layout per ISA
                #pragma unroll
                for (int t = 0; t < 9; ++t) {
                    const int dy = t / 3, dx = t % 3;
                    const _Float16* ap = s_a1 + (a2r+dy)*A1_ROW_STRIDE
                                       + (colbase + nn + dx)*32 + kbA;
                    v8h alo = *(const v8h*)ap;
                    v8h ahi = *(const v8h*)(ap + 16);
                    v16h A;
                    #pragma unroll
                    for (int e = 0; e < 8; ++e) { A[e] = alo[e]; A[e+8] = ahi[e]; }
                    #pragma unroll
                    for (int n = 0; n < 4; ++n) {
                        const int o = n*16 + nn;
                        const _Float16* bp = s_w2 + t*2048 + o*32 + kbB;
                        v8h blo = *(const v8h*)bp;
                        v8h bhi = *(const v8h*)(bp + 16);
                        v16h B;
                        #pragma unroll
                        for (int e = 0; e < 8; ++e) { B[e] = blo[e]; B[e+8] = bhi[e]; }
                        acc[n] = __builtin_amdgcn_wmma_f32_16x16x32_f16(
                            false, A, false, B, (short)0, acc[n], false, false);
                    }
                }
                // D layout: lane -> N, vgpr r -> M = r + (hi?8:0). Bias+ReLU.
                #pragma unroll
                for (int n = 0; n < 4; ++n) {
                    const int o = n*16 + nn;
                    const float bias = s_b2[o];
                    #pragma unroll
                    for (int r = 0; r < 8; ++r) {
                        int md = r + (hi ? 8 : 0);
                        int sc = colbase + md + 1;
                        float vv = fmaxf(acc[n][r] + bias, 0.f);
                        s_a2[o*A2_CH_STRIDE + a2r*A2_W + sc] = (_Float16)vv;
                    }
                }
            } else {  // out-of-image rows must be zero for conv3 SAME padding
                #pragma unroll
                for (int n = 0; n < 4; ++n) {
                    const int o = n*16 + nn;
                    #pragma unroll
                    for (int r = 0; r < 8; ++r) {
                        int md = r + (hi ? 8 : 0);
                        s_a2[o*A2_CH_STRIDE + a2r*A2_W + colbase + md + 1] = (_Float16)0.f;
                    }
                }
            }
        }
        __syncthreads();

        // ---------- conv3 + linear accumulate (fused, conv3 never stored) ----
        for (int i = tid; i < TILE_H*IMG; i += 256) {   // 768 px (uniform)
            int lr = i >> 7, c = i & 127;
            int gr = r0 + lr;
            float a3 = cb3;
            #pragma unroll 4
            for (int ch = 0; ch < 64; ++ch) {
                const _Float16* base = s_a2 + ch*A2_CH_STRIDE;
                const float* wp = s_w3 + ch*9;
                #pragma unroll
                for (int dy = 0; dy < 3; ++dy) {
                    const _Float16* rp = base + (lr+dy)*A2_W + c;
                    #pragma unroll
                    for (int dx = 0; dx < 3; ++dx)
                        a3 = fmaf((float)rp[dx], wp[dy*3+dx], a3);
                }
            }
            int sflat = gr*IMG + c;
            const float* lp = linw + sflat;     // lin_w stays hot in 192MB L2
            #pragma unroll
            for (int q = 0; q < 25; ++q)
                linacc[q] = fmaf(a3, lp[q*NPIX], linacc[q]);
        }
        __syncthreads();
    }

    // ---- deterministic reduction: wave shuffle, then cross-wave in LDS ----
    #pragma unroll
    for (int q = 0; q < 25; ++q) {
        float v = linacc[q];
        #pragma unroll
        for (int off = 16; off > 0; off >>= 1)
            v += __shfl_down(v, off, 32);
        if (lane == 0) s_red[wave*25 + q] = v;
    }
    __syncthreads();
    if (tid < 25) {
        float s = linb[tid];
        #pragma unroll
        for (int w = 0; w < 8; ++w) s += s_red[w*25 + tid];
        patches[p*25 + tid] = s;
    }
}

// ---------------------------------------------------------------------------
// Patch grid tiles exactly (stride==patch==5): cnt==1 on [0,125)^2, 0 outside.
// ---------------------------------------------------------------------------
__global__ void recon_sigmoid(const float* __restrict__ patches,
                              float* __restrict__ out)
{
    int idx = blockIdx.x*256 + threadIdx.x;
    if (idx >= IMG*IMG) return;
    int y = idx >> 7, x = idx & 127;
    float o;
    if (y < 125 && x < 125) {
        int p = (y/5)*25 + (x/5);
        int q = (y%5)*5 + (x%5);
        float r = patches[p*25 + q];
        o = 1.f / (1.f + expf(-r));
    } else {
        o = 0.5f;   // sigmoid(0) where cnt==0
    }
    out[idx] = o;
}

__device__ __forceinline__ float getpix(const float* img, int xi, int yi)
{
    bool v = (xi >= 0) & (xi < IMG) & (yi >= 0) & (yi < IMG);
    int xc = xi < 0 ? 0 : (xi > IMG-1 ? IMG-1 : xi);
    int yc = yi < 0 ? 0 : (yi > IMG-1 ? IMG-1 : yi);
    float val = img[yc*IMG + xc];
    return v ? val : 0.f;
}

__global__ void radon_k(const float* __restrict__ img, float* __restrict__ out)
{
    int a  = blockIdx.x;     // angle
    int tt = threadIdx.x;    // detector index
    float ang = (float)a * 0.017453292519943295f;
    float st, ct;
    sincosf(ang, &st, &ct);
    const float c0 = 63.5f;
    float tv = (float)tt - c0;
    float sum = 0.f;
    for (int ss = 0; ss < IMG; ++ss) {
        float sv = (float)ss - c0;
        float x = c0 + tv*ct - sv*st;
        float y = c0 + tv*st + sv*ct;
        float fx = floorf(x), fy = floorf(y);
        int x0 = (int)fx, y0 = (int)fy;
        float wx = x - fx, wy = y - fy;
        sum += getpix(img, x0,   y0  ) * (1.f-wx)*(1.f-wy)
             + getpix(img, x0+1, y0  ) * wx      *(1.f-wy)
             + getpix(img, x0,   y0+1) * (1.f-wx)*wy
             + getpix(img, x0+1, y0+1) * wx      *wy;
    }
    out[IMG*IMG + a*IMG + tt] = sum;
}

extern "C" void kernel_launch(void* const* d_in, const int* in_sizes, int n_in,
                              void* d_out, int out_size, void* d_ws, size_t ws_size,
                              hipStream_t stream)
{
    const float* sino = (const float*)d_in[0];
    const float* w1   = (const float*)d_in[1];
    const float* b1   = (const float*)d_in[2];
    const float* w2   = (const float*)d_in[3];
    const float* b2   = (const float*)d_in[4];
    const float* w3   = (const float*)d_in[5];
    const float* b3   = (const float*)d_in[6];
    const float* linw = (const float*)d_in[7];
    const float* linb = (const float*)d_in[8];
    const int*   masks= (const int*)d_in[9];
    // d_in[10] = starts: unused (exact tiling structure exploited)

    float* patches = (float*)d_ws;          // 625*25 f32 = 62.5 KB scratch
    float* out     = (float*)d_out;         // [16384 y_hat | 7680 s_hat]

    fused_batch<<<NPATCH, 256, SMEM_BYTES, stream>>>(
        sino, w1, b1, w2, b2, w3, b3, linw, linb, masks, patches);
    recon_sigmoid<<<(IMG*IMG + 255)/256, 256, 0, stream>>>(patches, out);
    radon_k<<<NANG, IMG, 0, stream>>>(out, out);
}
